// Variance_Block_28174985462486
// MI455X (gfx1250) — matrix operand order repaired and checked by reading, not compiled, not run on hardware
//
#include <hip/hip_runtime.h>
#include <hip/hip_bf16.h>

typedef __attribute__((ext_vector_type(16))) _Float16 v16h;
typedef __attribute__((ext_vector_type(8)))  float    v8f;
typedef __attribute__((ext_vector_type(4)))  float    v4f;
typedef __attribute__((ext_vector_type(4)))  int      v4i;

#define T_TEXT 512
#define CHUNKS 32              // T_TEXT / 16
#define ROWS_PER_BLOCK 32

// ---------------------------------------------------------------------------
// Kernel A: one wave32 per batch row.
//   repeats = (int)(d + 0.5)  (exact in f16, values 1..8)
//   Intra-chunk inclusive prefix sums via V_WMMA_F32_16X16X32_F16 with
//   B = 16x16 upper-triangular ones (K rows 16..31 zeroed). f32 accumulation
//   is exact (sums <= 4096). Then a 32-chunk offset scan; emit int starts/ends.
// ---------------------------------------------------------------------------
__global__ __launch_bounds__(32)
void lr_scan_wmma(const float* __restrict__ dur,
                  int* __restrict__ wsStarts,
                  int* __restrict__ wsEnds) {
  const int b    = blockIdx.x;
  const int lane = threadIdx.x;            // 0..31
  __shared__ float P[T_TEXT];              // intra-chunk inclusive prefixes

  const float* drow = dur + (size_t)b * T_TEXT;

  // B fragment: B[k][j] = (k <= j) for k<16, zero for k>=16.
  // Layout: lane l holds column N=l%16; elems e<8 hold K=e+8*(l/16), elems
  // e>=8 hold K=16..31 (all zero here).
  v16h bm;
#pragma unroll
  for (int e = 0; e < 16; ++e) {
    float one = (e < 8 && e + 8 * (lane >> 4) <= (lane & 15)) ? 1.0f : 0.0f;
    bm[e] = (_Float16)one;
  }

  // Two WMMAs cover chunk groups [0,16) and [16,32).
#pragma unroll
  for (int g = 0; g < 2; ++g) {
    const int m     = (lane & 15) + g * 16;   // chunk (= A-matrix row M) fed by this lane
    const int kbase = (lane < 16) ? 0 : 8;    // K-half per A layout
    v16h a;
#pragma unroll
    for (int e = 0; e < 8; ++e) {
      float d = drow[m * 16 + kbase + e];
      int   r = (int)(d + 0.5f);              // matches reference rounding (d>0)
      a[e]     = (_Float16)(float)r;          // K = kbase+e
      a[e + 8] = (_Float16)0.0f;              // K = 16..31 zero-pad
    }
    v8f c = {};
    // emits v_wmma_f32_16x16x32_f16
    v8f dacc = __builtin_amdgcn_wmma_f32_16x16x32_f16(
        /*neg_a=*/false, a, /*neg_b=*/false, bm,
        /*c_mod=*/(short)0, c, /*reuse_a=*/false, /*reuse_b=*/false);

    // D layout: elem v -> row M=v (lanes 0-15) / M=v+8 (lanes 16-31), col N=lane%16
    const int jcol = lane & 15;
    const int cb   = ((lane < 16) ? 0 : 8) + g * 16;
#pragma unroll
    for (int v = 0; v < 8; ++v)
      P[(cb + v) * 16 + jcol] = dacc[v];
  }
  __syncthreads();

  // Exclusive scan of chunk totals; lane c owns chunk c.
  float off = 0.0f;
  for (int c2 = 0; c2 < lane; ++c2) off += P[c2 * 16 + 15];

  const int c = lane;
  int sOut[16], eOut[16];
  float prev = 0.0f;
#pragma unroll
  for (int j = 0; j < 16; ++j) {
    float pj = P[c * 16 + j];
    eOut[j] = (int)(off + pj + 0.5f);
    sOut[j] = (int)(off + prev + 0.5f);
    prev = pj;
  }

  const int base = b * T_TEXT + c * 16;
#pragma unroll
  for (int q = 0; q < 4; ++q) {
    v4i sv = { sOut[4*q+0], sOut[4*q+1], sOut[4*q+2], sOut[4*q+3] };
    v4i ev = { eOut[4*q+0], eOut[4*q+1], eOut[4*q+2], eOut[4*q+3] };
    *(v4i*)(wsStarts + base + 4*q) = sv;
    *(v4i*)(wsEnds   + base + 4*q) = ev;
  }
}

// ---------------------------------------------------------------------------
// Kernel B: store-bandwidth-bound expansion. One block = one batch x 32 rows
// of t. Each thread owns 4 fixed columns (s/e held in registers across the
// whole t loop) and streams 16 x 16B nontemporal stores (output 256MB > L2,
// single consumer -> NT is the right temporal hint).
// ---------------------------------------------------------------------------
__global__ __launch_bounds__(256)
void lr_expand(const int* __restrict__ wsStarts,
               const int* __restrict__ wsEnds,
               float* __restrict__ out,
               int maxLen, int blocksPerBatch) {
  const int b    = blockIdx.x / blocksPerBatch;
  const int tile = blockIdx.x % blocksPerBatch;
  const int tid  = threadIdx.x;
  const int col4 = (tid & 127) * 4;   // 128 float4 groups cover T_TEXT=512
  const int rsub = tid >> 7;          // two t-rows in flight per pass

  const v4i s = *(const v4i*)(wsStarts + b * T_TEXT + col4);
  const v4i e = *(const v4i*)(wsEnds   + b * T_TEXT + col4);

  const int tBase = tile * ROWS_PER_BLOCK;
  float* orow = out + ((size_t)b * maxLen + tBase) * T_TEXT + col4;

#pragma unroll
  for (int i = 0; i < ROWS_PER_BLOCK / 2; ++i) {
    const int tOff = rsub + 2 * i;
    const int t    = tBase + tOff;
    if (t < maxLen) {
      v4f val;
      val.x = (t >= s.x && t < e.x) ? 1.0f : 0.0f;
      val.y = (t >= s.y && t < e.y) ? 1.0f : 0.0f;
      val.z = (t >= s.z && t < e.z) ? 1.0f : 0.0f;
      val.w = (t >= s.w && t < e.w) ? 1.0f : 0.0f;
      __builtin_nontemporal_store(val, (v4f*)(orow + (size_t)tOff * T_TEXT));
    }
  }
}

// ---------------------------------------------------------------------------
extern "C" void kernel_launch(void* const* d_in, const int* in_sizes, int n_in,
                              void* d_out, int out_size, void* d_ws, size_t ws_size,
                              hipStream_t stream) {
  const float* durations = (const float*)d_in[0];

  const int inTot  = in_sizes[0];          // B * T_TEXT
  const int B      = inTot / T_TEXT;       // 32
  const int maxLen = out_size / inTot;     // (B*ML*T)/(B*T) = 4096

  int* wsS = (int*)d_ws;
  int* wsE = wsS + (size_t)B * T_TEXT;     // 2 * B * 512 * 4B = 128 KB scratch

  lr_scan_wmma<<<B, 32, 0, stream>>>(durations, wsS, wsE);

  const int bpb = (maxLen + ROWS_PER_BLOCK - 1) / ROWS_PER_BLOCK;
  lr_expand<<<B * bpb, 256, 0, stream>>>(wsS, wsE, (float*)d_out, maxLen, bpb);
}
